// WindowAttention_39977555591563
// MI455X (gfx1250) — matrix operand order repaired and checked by reading, not compiled, not run on hardware
//
#include <hip/hip_runtime.h>

typedef __attribute__((ext_vector_type(16))) _Float16 v16h;
typedef __attribute__((ext_vector_type(8)))  _Float16 v8h;
typedef __attribute__((ext_vector_type(4)))  _Float16 v4h;
typedef __attribute__((ext_vector_type(8)))  float    v8f;

#define DIMC   512
#define NHEAD  16
#define NTOK   49
#define NN2    2401            // 49*49
#define SCALE_F 0.17677669529663687f   // (512/16)^-0.5

// ---- LDS layout (f16 element offsets) ----
#define XS_OFF 0               // x / attn-out scratch: 49x512
#define QS_OFF 25088           // Q|K: 49x1024  (cols [0,512)=Q, [512,1024)=K)
#define VT_OFF 75264           // V transposed: [16 heads][32 d][64 tok]
#define PS_OFF 108032          // P scratch: 16 waves x (16x64)
#define LDS_ELEMS 124416
#define LDS_BYTES (LDS_ELEMS * 2)

// ---- WS layout (bytes) ----
#define WS_WQKV  0             // 1536*512 f16
#define WS_WPROJ 1572864       // 512*512 f16
#define WS_COMB  2097152       // 64*16*49*49 f32 (rel-pos bias + shift mask, combined)

// A-operand fragment (16x32 f16): lane-half h -> K chunks [8h,8h+8) and [16+8h,16+8h+8)
static __device__ __forceinline__ v16h frag_a(const _Float16* p, int lh) {
    v8h lo = *(const v8h*)(p + 8 * lh);
    v8h hi = *(const v8h*)(p + 16 + 8 * lh);
    return __builtin_shufflevector(lo, hi, 0,1,2,3,4,5,6,7,8,9,10,11,12,13,14,15);
}
// B-operand fragment (32x16 f16): lane-half h -> K chunks [8h,8h+8) and [16+8h,16+8h+8)
static __device__ __forceinline__ v16h frag_b(const _Float16* p, int lh) {
    v8h lo = *(const v8h*)(p + 8 * lh);
    v8h hi = *(const v8h*)(p + 16 + 8 * lh);
    return __builtin_shufflevector(lo, hi, 0,1,2,3,4,5,6,7,8,9,10,11,12,13,14,15);
}
static __device__ __forceinline__ v8f wmma16(v16h a, v16h b, v8f c) {
    return __builtin_amdgcn_wmma_f32_16x16x32_f16(false, a, false, b, (short)0, c, false, false);
}

// ---------------- prep kernels ----------------
__global__ __launch_bounds__(256) void cvt_f32_to_f16(const float* __restrict__ s,
                                                      _Float16* __restrict__ d, int n) {
    int i = blockIdx.x * 256 + threadIdx.x;
    if (i < n) d[i] = (_Float16)s[i];
}

// comb[wi][h][m][n] = rpb_table[rel_idx[m*49+n]*16 + h] + mask[wi][m][n]
__global__ __launch_bounds__(256) void build_comb(const float* __restrict__ rpb,
                                                  const int* __restrict__ rel,
                                                  const float* __restrict__ mask,
                                                  float* __restrict__ comb) {
    int i = blockIdx.x * 256 + threadIdx.x;
    if (i < 64 * NHEAD * NN2) {
        int wi  = i / (NHEAD * NN2);
        int rem = i - wi * (NHEAD * NN2);
        int h   = rem / NN2;
        int mn  = rem - h * NN2;
        comb[i] = rpb[rel[mn] * NHEAD + h] + mask[wi * NN2 + mn];
    }
}

// ---------------- fused per-window kernel ----------------
__global__ __launch_bounds__(512, 1)
void fused_window_attn(const float* __restrict__ x,
                       const float* __restrict__ qkv_b,
                       const float* __restrict__ proj_b,
                       const _Float16* __restrict__ wqkv,
                       const _Float16* __restrict__ wproj,
                       const float* __restrict__ comb,
                       float* __restrict__ out) {
    extern __shared__ _Float16 smem[];
    _Float16* xs = smem + XS_OFF;   // 49x512 (x, later attn-out)
    _Float16* qs = smem + QS_OFF;   // 49x1024
    _Float16* vt = smem + VT_OFF;   // 16x32x64
    _Float16* ps = smem + PS_OFF;   // per-wave 16x64

    const int w    = blockIdx.x;
    const int tid  = threadIdx.x;
    const int lane = tid & 31;
    const int wave = tid >> 5;          // 0..15
    const int l16  = lane & 15;
    const int lh   = lane >> 4;

    // ---- phase 0/1: zero Vt padding (v8h), stage x as f16 (float4-wide) ----
    {
        const v8h z = {};
        for (int i = tid; i < (NHEAD * 32 * 64) / 8; i += 512) *(v8h*)(vt + i * 8) = z;
        const float4* xw4 = (const float4*)(x + (size_t)w * (NTOK * DIMC));
        for (int i = tid; i < (NTOK * DIMC) / 4; i += 512) {
            float4 v = xw4[i];
            v4h hv = { (_Float16)v.x, (_Float16)v.y, (_Float16)v.z, (_Float16)v.w };
            *(v4h*)(xs + i * 4) = hv;
        }
    }
    __syncthreads();

    // ---- phase 2: QKV GEMM (49x512 @ 512x1536) ----
    // k-outer, 4(m) x 3(n) accumulator block; each wave owns 6 n-tiles as 2 groups of 3.
    {
        const _Float16* arow[4];
        #pragma unroll
        for (int mt = 0; mt < 4; ++mt) {
            int mrow = mt * 16 + l16; if (mrow > 48) mrow = 48;
            arow[mt] = xs + mrow * DIMC;
        }
        #pragma unroll
        for (int g = 0; g < 2; ++g) {
            const int ntb = wave * 6 + g * 3;
            int o[3]; float bo[3]; const _Float16* brow[3];
            #pragma unroll
            for (int j = 0; j < 3; ++j) {
                o[j] = (ntb + j) * 16 + l16;
                bo[j] = qkv_b[o[j]];
                brow[j] = wqkv + (size_t)o[j] * DIMC;
            }
            v8f acc[4][3] = {};
            #pragma unroll
            for (int ks = 0; ks < 16; ++ks) {
                v16h A[4];
                #pragma unroll
                for (int mt = 0; mt < 4; ++mt) A[mt] = frag_a(arow[mt] + ks * 32, lh);
                #pragma unroll
                for (int j = 0; j < 3; ++j) {
                    v16h B = frag_b(brow[j] + ks * 32, lh);
                    #pragma unroll
                    for (int mt = 0; mt < 4; ++mt) acc[mt][j] = wmma16(A[mt], B, acc[mt][j]);
                }
            }
            #pragma unroll
            for (int j = 0; j < 3; ++j) {
                if (o[j] < 1024) {                   // Q or K -> row-major LDS
                    #pragma unroll
                    for (int mt = 0; mt < 4; ++mt)
                        #pragma unroll
                        for (int r = 0; r < 8; ++r) {
                            int tok = mt * 16 + r + 8 * lh;
                            if (tok < NTOK) qs[tok * 1024 + o[j]] = (_Float16)(acc[mt][j][r] + bo[j]);
                        }
                } else {                             // V -> transposed LDS [h][d][tok]
                    int dg = o[j] - 1024, hh = dg >> 5, dd = dg & 31;
                    #pragma unroll
                    for (int mt = 0; mt < 4; ++mt)
                        #pragma unroll
                        for (int r = 0; r < 8; ++r) {
                            int tok = mt * 16 + r + 8 * lh;
                            if (tok < NTOK) vt[hh * 2048 + dd * 64 + tok] = (_Float16)(acc[mt][j][r] + bo[j]);
                        }
                }
            }
        }
    }
    __syncthreads();

    // ---- phase 3: attention, one head per wave ----
    {
        const int h = wave;
        v16h aQ[4], bK[4];
        #pragma unroll
        for (int mt = 0; mt < 4; ++mt) {
            int mrow = mt * 16 + l16; if (mrow > 48) mrow = 48;
            aQ[mt] = frag_a(qs + mrow * 1024 + h * 32, lh);
        }
        #pragma unroll
        for (int nt = 0; nt < 4; ++nt) {
            int tok = nt * 16 + l16; if (tok > 48) tok = 48;
            bK[nt] = frag_b(qs + tok * 1024 + 512 + h * 32, lh);
        }
        v8f S[4][4];
        #pragma unroll
        for (int mt = 0; mt < 4; ++mt)
            #pragma unroll
            for (int nt = 0; nt < 4; ++nt) {
                v8f z = {};
                S[mt][nt] = wmma16(aQ[mt], bK[nt], z);
            }

        const float* cb = comb + (size_t)((w & 63) * NHEAD + h) * NN2;
        _Float16* pw = ps + wave * 1024;

        #pragma unroll
        for (int mt = 0; mt < 4; ++mt) {
            // scale + (bias+mask) combined table (invalid cols -> -1e30)
            #pragma unroll
            for (int nt = 0; nt < 4; ++nt) {
                const int n = nt * 16 + l16;
                const bool valid = (n < NTOK);
                #pragma unroll
                for (int r = 0; r < 8; ++r) {
                    int m = mt * 16 + r + 8 * lh; if (m > 48) m = 48;
                    float sv = S[mt][nt][r] * SCALE_F;
                    sv += valid ? cb[m * NTOK + n] : -1e30f;
                    S[mt][nt][r] = sv;
                }
            }
            // row softmax: each row lives in one 16-lane half
            #pragma unroll
            for (int r = 0; r < 8; ++r) {
                float mx = fmaxf(fmaxf(S[mt][0][r], S[mt][1][r]),
                                 fmaxf(S[mt][2][r], S[mt][3][r]));
                #pragma unroll
                for (int sft = 1; sft < 16; sft <<= 1) mx = fmaxf(mx, __shfl_xor(mx, sft, 32));
                float p0 = __expf(S[mt][0][r] - mx);
                float p1 = __expf(S[mt][1][r] - mx);
                float p2 = __expf(S[mt][2][r] - mx);
                float p3 = __expf(S[mt][3][r] - mx);
                float sum = p0 + p1 + p2 + p3;
                #pragma unroll
                for (int sft = 1; sft < 16; sft <<= 1) sum += __shfl_xor(sum, sft, 32);
                const float inv = 1.0f / sum;
                const int mr = r + 8 * lh;
                pw[mr * 64 + 0  + l16] = (_Float16)(p0 * inv);
                pw[mr * 64 + 16 + l16] = (_Float16)(p1 * inv);
                pw[mr * 64 + 32 + l16] = (_Float16)(p2 * inv);
                pw[mr * 64 + 48 + l16] = (_Float16)(p3 * inv);
            }
            // P @ V  (contraction over 64 padded tokens = 2 WMMA k-steps)
            v16h aP0 = frag_a(pw + l16 * 64, lh);
            v16h aP1 = frag_a(pw + l16 * 64 + 32, lh);
            #pragma unroll
            for (int dt = 0; dt < 2; ++dt) {
                const _Float16* vrow = vt + h * 2048 + (dt * 16 + l16) * 64;
                v8f o8 = {};
                o8 = wmma16(aP0, frag_b(vrow, lh), o8);
                o8 = wmma16(aP1, frag_b(vrow + 32, lh), o8);
                #pragma unroll
                for (int r = 0; r < 8; ++r) {
                    int tok = mt * 16 + r + 8 * lh;
                    if (tok < NTOK) xs[tok * DIMC + h * 32 + dt * 16 + l16] = (_Float16)o8[r];
                }
            }
        }
    }
    __syncthreads();

    // ---- phase 4: output projection (49x512 @ 512x512) + bias -> f32 out ----
    // k-outer, 4(m) x 2(n) accumulator block; each wave owns 2 n-tiles.
    {
        float* orow = out + (size_t)w * (NTOK * DIMC);
        const _Float16* arow[4];
        #pragma unroll
        for (int mt = 0; mt < 4; ++mt) {
            int mrow = mt * 16 + l16; if (mrow > 48) mrow = 48;
            arow[mt] = xs + mrow * DIMC;
        }
        int o[2]; float pb[2]; const _Float16* brow[2];
        #pragma unroll
        for (int t = 0; t < 2; ++t) {
            o[t] = (wave * 2 + t) * 16 + l16;
            pb[t] = proj_b[o[t]];
            brow[t] = wproj + (size_t)o[t] * DIMC;
        }
        v8f acc[4][2] = {};
        #pragma unroll
        for (int ks = 0; ks < 16; ++ks) {
            v16h A[4];
            #pragma unroll
            for (int mt = 0; mt < 4; ++mt) A[mt] = frag_a(arow[mt] + ks * 32, lh);
            #pragma unroll
            for (int t = 0; t < 2; ++t) {
                v16h B = frag_b(brow[t] + ks * 32, lh);
                #pragma unroll
                for (int mt = 0; mt < 4; ++mt) acc[mt][t] = wmma16(A[mt], B, acc[mt][t]);
            }
        }
        #pragma unroll
        for (int t = 0; t < 2; ++t)
            #pragma unroll
            for (int mt = 0; mt < 4; ++mt)
                #pragma unroll
                for (int r = 0; r < 8; ++r) {
                    int tok = mt * 16 + r + 8 * lh;
                    if (tok < NTOK) orow[tok * DIMC + o[t]] = acc[mt][t][r] + pb[t];
                }
    }
}

extern "C" void kernel_launch(void* const* d_in, const int* in_sizes, int n_in,
                              void* d_out, int out_size, void* d_ws, size_t ws_size,
                              hipStream_t stream) {
    (void)in_sizes; (void)n_in; (void)out_size; (void)ws_size;
    const float* x      = (const float*)d_in[0];
    const float* mask   = (const float*)d_in[1];
    const float* qkv_w  = (const float*)d_in[2];
    const float* qkv_b  = (const float*)d_in[3];
    const float* proj_w = (const float*)d_in[4];
    const float* proj_b = (const float*)d_in[5];
    const float* rpb    = (const float*)d_in[6];
    const int*   rel    = (const int*)d_in[7];

    char* ws = (char*)d_ws;
    _Float16* wqkv  = (_Float16*)(ws + WS_WQKV);
    _Float16* wproj = (_Float16*)(ws + WS_WPROJ);
    float*    comb  = (float*)(ws + WS_COMB);

    cvt_f32_to_f16<<<(1536 * 512 + 255) / 256, 256, 0, stream>>>(qkv_w, wqkv, 1536 * 512);
    cvt_f32_to_f16<<<(512 * 512 + 255) / 256, 256, 0, stream>>>(proj_w, wproj, 512 * 512);
    build_comb<<<(64 * NHEAD * NN2 + 255) / 256, 256, 0, stream>>>(rpb, rel, mask, comb);

    (void)hipFuncSetAttribute((const void*)fused_window_attn,
                              hipFuncAttributeMaxDynamicSharedMemorySize, LDS_BYTES);
    fused_window_attn<<<2048, 512, LDS_BYTES, stream>>>(
        x, qkv_b, proj_b, wqkv, wproj, comb, (float*)d_out);
}
